// Attention_49383533969427
// MI455X (gfx1250) — matrix operand order repaired and checked by reading, not compiled, not run on hardware
//
#include <hip/hip_runtime.h>
#include <hip/hip_bf16.h>

typedef __attribute__((ext_vector_type(16))) _Float16 v16h;
typedef __attribute__((ext_vector_type(8)))  _Float16 half8;
typedef __attribute__((ext_vector_type(8)))  float    v8f;

#define EMBED   768
#define NHEAD   12
#define HDIM    64
#define SEQ     2048
#define BATCH   4
#define ROWS    (BATCH * SEQ)     /* 8192 */
#define QKV_OUT (3 * EMBED)       /* 2304 */
#define BH      (BATCH * NHEAD)   /* 48   */

union V16U { v16h v; half8 h[2]; };

// ---------------------------------------------------------------------------
// WMMA helper: D = A(16x32 f16) * B(32x16 f16) + C(16x16 f32)
// ---------------------------------------------------------------------------
__device__ __forceinline__ v8f wmma_f16(v16h a, v16h b, v8f c) {
    return __builtin_amdgcn_wmma_f32_16x16x32_f16(
        /*neg_a=*/false, a, /*neg_b=*/false, b,
        /*c_mod=*/(short)0, c, /*reuse_a=*/false, /*reuse_b=*/false);
}

// A-fragment (16x32, f16): lane L<16 holds row L with K = k..k+7 (halves 0-7)
// and K = k+16..k+23 (halves 8-15); lane L+16 holds K = k+8.. / k+24..
// `row` is the per-lane row index; source is row-major with leading dim `ld`.
__device__ __forceinline__ v16h load_afrag(const _Float16* __restrict__ base,
                                           int ld, int row, int k, int lane) {
    const int hi = (lane >> 4) & 1;
    const _Float16* p = base + (size_t)row * ld + k + hi * 8;
    V16U u;
    u.h[0] = *(const half8*)(p);
    u.h[1] = *(const half8*)(p + 16);
    return u.v;
}

// B-fragment (32x16, f16): lane n<16 holds column n with K = k..k+7 and
// K = k+16..k+23; lane n+16 holds K = k+8.. / k+24..  A column of B is a row of
// the row-major source (leading dim `ld`), hence contiguous per lane.
__device__ __forceinline__ v16h load_bfrag(const _Float16* __restrict__ base,
                                           int ld, int col, int k, int lane) {
    const int hi = (lane >> 4) & 1;
    const _Float16* p = base + (size_t)col * ld + k + hi * 8;
    V16U u;
    u.h[0] = *(const half8*)(p);
    u.h[1] = *(const half8*)(p + 16);
    return u.v;
}

// ---------------------------------------------------------------------------
// Kernel 0: fp32 -> f16 conversion
// ---------------------------------------------------------------------------
__global__ void f32_to_f16_kernel(const float* __restrict__ in,
                                  _Float16* __restrict__ out, int n) {
    int i = blockIdx.x * blockDim.x + threadIdx.x;
    if (i < n) out[i] = (_Float16)in[i];
}

// ---------------------------------------------------------------------------
// Kernel 1: QKV GEMM.  X[8192,768] * Wqkv^T[768,2304] -> Q,K scaled/layouted
// as [BH, SEQ, HDIM] f16, V transposed as [BH, HDIM, SEQ] f16.
// One wave computes a 64x64 output tile (4x4 WMMA accumulators).
// ---------------------------------------------------------------------------
__global__ __launch_bounds__(256)
void qkv_gemm_kernel(const _Float16* __restrict__ X,
                     const _Float16* __restrict__ W,
                     _Float16* __restrict__ Qd,
                     _Float16* __restrict__ Kd,
                     _Float16* __restrict__ Vt) {
    const int lane = threadIdx.x & 31;
    const int wave = blockIdx.x * (blockDim.x >> 5) + (threadIdx.x >> 5);
    const int TN = QKV_OUT / 64;              // 36 tiles along output features
    const int TM = ROWS / 64;                 // 128 tiles along tokens
    const int tm = wave / TN;
    const int tn = wave % TN;
    if (tm >= TM) return;

    const int row0 = tm * 64;
    const int col0 = tn * 64;
    const int hi = lane >> 4;
    const int ln = lane & 15;

    v8f acc[4][4];
    const v8f vzero = {};
    #pragma unroll
    for (int m = 0; m < 4; ++m)
        #pragma unroll
        for (int n = 0; n < 4; ++n) acc[m][n] = vzero;

    for (int k = 0; k < EMBED; k += 32) {
        v16h a[4], b[4];
        #pragma unroll
        for (int m = 0; m < 4; ++m)
            a[m] = load_afrag(X, EMBED, row0 + m * 16 + ln, k, lane);
        #pragma unroll
        for (int n = 0; n < 4; ++n)
            b[n] = load_bfrag(W, EMBED, col0 + n * 16 + ln, k, lane);
        #pragma unroll
        for (int m = 0; m < 4; ++m)
            #pragma unroll
            for (int n = 0; n < 4; ++n)
                acc[m][n] = wmma_f16(a[m], b[n], acc[m][n]);
    }

    const float scale = 0.125f;  // HDIM^-0.5
    #pragma unroll
    for (int n = 0; n < 4; ++n) {
        const int obase  = col0 + n * 16;        // 16-aligned; single matrix/head
        const int mat    = obase / EMBED;        // 0=Q 1=K 2=V
        const int within = obase % EMBED;
        const int h      = within / HDIM;
        const int d      = (within % HDIM) + ln; // per-lane feature within head
        #pragma unroll
        for (int m = 0; m < 4; ++m) {
            const int trow0 = row0 + m * 16 + hi * 8;  // 8 consecutive tokens
            const int b_    = trow0 / SEQ;
            const int tok0  = trow0 % SEQ;
            const int bh    = b_ * NHEAD + h;
            if (mat == 2) {
                // V transposed: Vt[bh][d][token]; 8 consecutive tokens -> b128
                half8 pk;
                #pragma unroll
                for (int r = 0; r < 8; ++r) pk[r] = (_Float16)acc[m][n][r];
                *(half8*)(Vt + ((size_t)bh * HDIM + d) * SEQ + tok0) = pk;
            } else {
                _Float16* dst = (mat == 0) ? Qd : Kd;
                const float s = (mat == 0) ? scale : 1.0f;
                #pragma unroll
                for (int r = 0; r < 8; ++r)
                    dst[((size_t)bh * SEQ + tok0 + r) * HDIM + d] =
                        (_Float16)(acc[m][n][r] * s);
            }
        }
    }
}

// ---------------------------------------------------------------------------
// Kernel 2: flash attention.  One wave owns a 16-row Q tile of one (b,h),
// streams K/V in 32-column chunks with online softmax.
// S layout (C-matrix): VGPR r holds row r (lanes 0-15) / row r+8 (lanes 16-31),
// lane&15 = column -> row reductions are butterflies over the 16-lane half.
// ---------------------------------------------------------------------------
__global__ __launch_bounds__(256)
void flash_attn_kernel(const _Float16* __restrict__ Qd,
                       const _Float16* __restrict__ Kd,
                       const _Float16* __restrict__ Vt,
                       _Float16* __restrict__ Oh) {
    __shared__ _Float16 lds[8][16 * 32];   // per-wave P staging (1 KB each)
    const int lane  = threadIdx.x & 31;
    const int wslot = threadIdx.x >> 5;
    const int wave  = blockIdx.x * 8 + wslot;
    const int qt    = wave % (SEQ / 16);   // 128 row tiles per (b,h)
    const int bh    = wave / (SEQ / 16);
    const int hi = lane >> 4;
    const int ln = lane & 15;

    const _Float16* Qb = Qd + (size_t)bh * SEQ * HDIM;
    const _Float16* Kb = Kd + (size_t)bh * SEQ * HDIM;
    const _Float16* Vb = Vt + (size_t)bh * HDIM * SEQ;

    // Q fragments for the whole 16x64 tile stay resident (K chunks 0-31,32-63)
    v16h qf0 = load_afrag(Qb, HDIM, qt * 16 + ln, 0, lane);
    v16h qf1 = load_afrag(Qb, HDIM, qt * 16 + ln, 32, lane);

    const v8f vzero = {};
    v8f o_acc[4];
    #pragma unroll
    for (int t = 0; t < 4; ++t) o_acc[t] = vzero;
    float m_run[8], l_run[8];
    #pragma unroll
    for (int r = 0; r < 8; ++r) { m_run[r] = -1.0e30f; l_run[r] = 0.0f; }

    _Float16* P = lds[wslot];

    for (int j = 0; j < SEQ; j += 32) {
        if (j + 32 < SEQ) {   // gfx1250 global_prefetch_b8 for next K/V tiles
            __builtin_prefetch(Kb + (size_t)(j + 32) * HDIM, 0, 1);
            __builtin_prefetch(Vb + (size_t)(ln * 16) * SEQ + j + 32, 0, 1);
        }
        // ---- S = Q * K^T over this 32-column chunk (two 16x16 C tiles) ----
        v8f s0 = vzero, s1 = vzero;
        {
            v16h b00 = load_bfrag(Kb, HDIM, j + ln,      0,  lane);
            v16h b01 = load_bfrag(Kb, HDIM, j + ln,      32, lane);
            s0 = wmma_f16(qf0, b00, s0);
            s0 = wmma_f16(qf1, b01, s0);
            v16h b10 = load_bfrag(Kb, HDIM, j + 16 + ln, 0,  lane);
            v16h b11 = load_bfrag(Kb, HDIM, j + 16 + ln, 32, lane);
            s1 = wmma_f16(qf0, b10, s1);
            s1 = wmma_f16(qf1, b11, s1);
        }
        // ---- online softmax ----
        float alpha[8];
        #pragma unroll
        for (int r = 0; r < 8; ++r) {
            float mx = fmaxf(s0[r], s1[r]);
            #pragma unroll
            for (int off = 1; off < 16; off <<= 1)
                mx = fmaxf(mx, __shfl_xor(mx, off, 32));
            const float mnew = fmaxf(m_run[r], mx);
            alpha[r] = __expf(m_run[r] - mnew);
            m_run[r] = mnew;
            const float p0 = __expf(s0[r] - mnew);
            const float p1 = __expf(s1[r] - mnew);
            s0[r] = p0; s1[r] = p1;
            float rs = p0 + p1;
            #pragma unroll
            for (int off = 1; off < 16; off <<= 1)
                rs += __shfl_xor(rs, off, 32);
            l_run[r] = l_run[r] * alpha[r] + rs;
        }
        #pragma unroll
        for (int t = 0; t < 4; ++t)
            #pragma unroll
            for (int r = 0; r < 8; ++r) o_acc[t][r] *= alpha[r];

        // ---- stage P (C-layout f32 -> row-major f16 in LDS -> A-layout) ----
        __syncthreads();
        #pragma unroll
        for (int r = 0; r < 8; ++r) {
            const int row = r + hi * 8;
            P[row * 32 + ln]      = (_Float16)s0[r];
            P[row * 32 + 16 + ln] = (_Float16)s1[r];
        }
        __syncthreads();
        v16h pf;
        {
            const int base = hi * 8;
            V16U u;
            u.h[0] = *(const half8*)(P + ln * 32 + base);
            u.h[1] = *(const half8*)(P + ln * 32 + base + 16);
            pf = u.v;
        }
        // ---- O += P * V  (V transposed -> contiguous B fragments) ----
        #pragma unroll
        for (int t = 0; t < 4; ++t) {
            v16h bv = load_bfrag(Vb, SEQ, t * 16 + ln, j, lane);
            o_acc[t] = wmma_f16(pf, bv, o_acc[t]);
        }
    }

    // ---- epilogue: normalize and write O back token-major [B,N,C] ----
    const int tok0 = qt * 16 + hi * 8;
    const int b_ = bh / NHEAD;
    const int h  = bh % NHEAD;
    #pragma unroll
    for (int t = 0; t < 4; ++t) {
        const int d = t * 16 + ln;
        #pragma unroll
        for (int r = 0; r < 8; ++r) {
            const float v = o_acc[t][r] / l_run[r];
            Oh[((size_t)(b_ * SEQ + tok0 + r)) * EMBED + h * HDIM + d] =
                (_Float16)v;
        }
    }
}

// ---------------------------------------------------------------------------
// Kernel 3: output projection.  Oh[8192,768] * Wproj^T[768,768] + bias -> f32
// ---------------------------------------------------------------------------
__global__ __launch_bounds__(256)
void proj_gemm_kernel(const _Float16* __restrict__ Oh,
                      const _Float16* __restrict__ W,
                      const float* __restrict__ bias,
                      float* __restrict__ out) {
    const int lane = threadIdx.x & 31;
    const int wave = blockIdx.x * (blockDim.x >> 5) + (threadIdx.x >> 5);
    const int TN = EMBED / 64;   // 12
    const int TM = ROWS / 64;    // 128
    const int tm = wave / TN;
    const int tn = wave % TN;
    if (tm >= TM) return;

    const int row0 = tm * 64;
    const int col0 = tn * 64;
    const int hi = lane >> 4;
    const int ln = lane & 15;

    v8f acc[4][4];
    const v8f vzero = {};
    #pragma unroll
    for (int m = 0; m < 4; ++m)
        #pragma unroll
        for (int n = 0; n < 4; ++n) acc[m][n] = vzero;

    for (int k = 0; k < EMBED; k += 32) {
        v16h a[4], b[4];
        #pragma unroll
        for (int m = 0; m < 4; ++m)
            a[m] = load_afrag(Oh, EMBED, row0 + m * 16 + ln, k, lane);
        #pragma unroll
        for (int n = 0; n < 4; ++n)
            b[n] = load_bfrag(W, EMBED, col0 + n * 16 + ln, k, lane);
        #pragma unroll
        for (int m = 0; m < 4; ++m)
            #pragma unroll
            for (int n = 0; n < 4; ++n)
                acc[m][n] = wmma_f16(a[m], b[n], acc[m][n]);
    }

    #pragma unroll
    for (int n = 0; n < 4; ++n) {
        const int col = col0 + n * 16 + ln;
        const float bv = bias[col];
        #pragma unroll
        for (int m = 0; m < 4; ++m) {
            const int r0 = row0 + m * 16 + hi * 8;
            #pragma unroll
            for (int r = 0; r < 8; ++r)
                out[(size_t)(r0 + r) * EMBED + col] = acc[m][n][r] + bv;
        }
    }
}

// ---------------------------------------------------------------------------
// Host-side launcher
// ---------------------------------------------------------------------------
extern "C" void kernel_launch(void* const* d_in, const int* in_sizes, int n_in,
                              void* d_out, int out_size, void* d_ws, size_t ws_size,
                              hipStream_t stream) {
    (void)in_sizes; (void)n_in; (void)out_size; (void)ws_size;
    const float* x      = (const float*)d_in[0];   // [4,2048,768]
    const float* w_qkv  = (const float*)d_in[1];   // [2304,768]
    const float* w_proj = (const float*)d_in[2];   // [768,768]
    const float* b_proj = (const float*)d_in[3];   // [768]
    float* out = (float*)d_out;                    // [4,2048,768]

    _Float16* ws = (_Float16*)d_ws;
    size_t off = 0;
    _Float16* Xh  = ws + off; off += (size_t)ROWS * EMBED;      // 6.29M
    _Float16* Wqh = ws + off; off += (size_t)QKV_OUT * EMBED;   // 1.77M
    _Float16* Wph = ws + off; off += (size_t)EMBED * EMBED;     // 0.59M
    _Float16* Qd  = ws + off; off += (size_t)BH * SEQ * HDIM;   // 6.29M
    _Float16* Kd  = ws + off; off += (size_t)BH * SEQ * HDIM;   // 6.29M
    _Float16* Vt  = ws + off; off += (size_t)BH * HDIM * SEQ;   // 6.29M
    _Float16* Oh  = ws + off; off += (size_t)ROWS * EMBED;      // 6.29M  (~68MB)

    const int n_x = ROWS * EMBED;
    const int n_wq = QKV_OUT * EMBED;
    const int n_wp = EMBED * EMBED;
    f32_to_f16_kernel<<<(n_x  + 255) / 256, 256, 0, stream>>>(x,      Xh,  n_x);
    f32_to_f16_kernel<<<(n_wq + 255) / 256, 256, 0, stream>>>(w_qkv,  Wqh, n_wq);
    f32_to_f16_kernel<<<(n_wp + 255) / 256, 256, 0, stream>>>(w_proj, Wph, n_wp);

    // 128 x 36 = 4608 waves, 8 waves / block
    qkv_gemm_kernel<<<576, 256, 0, stream>>>(Xh, Wqh, Qd, Kd, Vt);
    // 48 bh * 128 row-tiles = 6144 waves, 8 waves / block
    flash_attn_kernel<<<768, 256, 0, stream>>>(Qd, Kd, Vt, Oh);
    // 128 x 12 = 1536 waves, 8 waves / block
    proj_gemm_kernel<<<192, 256, 0, stream>>>(Oh, Wph, b_proj, out);
}
